// HybridCodebook_65944927863113
// MI455X (gfx1250) — matrix operand order repaired and checked by reading, not compiled, not run on hardware
//
#include <hip/hip_runtime.h>

typedef __attribute__((ext_vector_type(16))) _Float16 v16h;
typedef __attribute__((ext_vector_type(8)))  _Float16 v8h;
typedef __attribute__((ext_vector_type(8)))  float    v8f;

#define D      1024
#define N_X    8192
#define N_SEM  8192
#define N_LRN  128
#define N_CB   8320   // N_SEM + N_LRN
#define EPSF   1e-8f

// ---------------------------------------------------------------------------
// Row-normalize rows of `src` [rows, 1024]; write f32 and f16 copies.
// ---------------------------------------------------------------------------
__global__ __launch_bounds__(256)
void nrm_kernel(const float* __restrict__ src, float* __restrict__ dstF,
                _Float16* __restrict__ dstH) {
  const int row = blockIdx.x;
  const int t   = threadIdx.x;
  const float* s = src + (size_t)row * D;

  __shared__ float red[256];
  float v[4];
  float ss = 0.f;
#pragma unroll
  for (int i = 0; i < 4; ++i) { v[i] = s[t + i * 256]; ss += v[i] * v[i]; }
  red[t] = ss;
  __syncthreads();
  for (int off = 128; off > 0; off >>= 1) {
    if (t < off) red[t] += red[t + off];
    __syncthreads();
  }
  const float inv = 1.f / fmaxf(sqrtf(red[0]), EPSF);

  float*    dF = dstF + (size_t)row * D;
  _Float16* dH = dstH + (size_t)row * D;
#pragma unroll
  for (int i = 0; i < 4; ++i) {
    const float nv = v[i] * inv;
    dF[t + i * 256] = nv;
    dH[t + i * 256] = (_Float16)nv;
  }
}

// ---------------------------------------------------------------------------
// logits[8192, 8320] = A[8192,1024] * B[8320,1024]^T   (f16 in, f32 out)
// One wave owns a 32(M) x 64(N) strip: 2x4 accumulators; each B fragment is
// reused by 2 A fragments, each A fragment by 4 B fragments (8 WMMA / K-step).
// Full unroll of the K-loop (32 steps, 256 WMMAs) so the backend can keep
// many global_load_b128 in flight instead of draining loadcnt per step.
// Fragment layouts used here (contiguous K order, same for A and B; not the ISA table order):
//   A 16x32 f16: lanes 0-15 -> M=lane, K=0..15;
//                lanes 16-31 -> M=lane-16, K=16..31
//   B 32x16 f16: lanes 0-15 -> N=lane, K=0..15; lanes 16-31 -> N=lane-16, K=16..31
// ---------------------------------------------------------------------------
__global__ __launch_bounds__(256)
void gemm_kernel(const _Float16* __restrict__ A,
                 const _Float16* __restrict__ B,
                 float* __restrict__ C) {
  const int lane   = threadIdx.x & 31;
  const int wave   = threadIdx.x >> 5;
  const int mTile  = (blockIdx.y * 8 + wave) * 32;
  const int nBase  = blockIdx.x * 64;
  const int laneHi = lane >> 4;   // 0 or 1
  const int lane15 = lane & 15;

  // Per-lane A bases: rows mTile+lane15 and mTile+16+lane15
  const _Float16* aRow0 = A + (size_t)(mTile + lane15) * D + 16 * laneHi;
  const _Float16* aRow1 = aRow0 + (size_t)16 * D;

  v8f acc[2][4];
#pragma unroll
  for (int mi = 0; mi < 2; ++mi)
#pragma unroll
    for (int ni = 0; ni < 4; ++ni) acc[mi][ni] = (v8f){};

#pragma unroll
  for (int k0 = 0; k0 < D; k0 += 32) {
    // A fragments: halves K=[base..base+7] and K=[base+8..base+15]
    v8h a0lo = *(const v8h*)(aRow0 + k0);
    v8h a0hi = *(const v8h*)(aRow0 + k0 + 8);
    v8h a1lo = *(const v8h*)(aRow1 + k0);
    v8h a1hi = *(const v8h*)(aRow1 + k0 + 8);
    v16h a0 = __builtin_shufflevector(a0lo, a0hi, 0, 1, 2, 3, 4, 5, 6, 7,
                                                  8, 9, 10, 11, 12, 13, 14, 15);
    v16h a1 = __builtin_shufflevector(a1lo, a1hi, 0, 1, 2, 3, 4, 5, 6, 7,
                                                  8, 9, 10, 11, 12, 13, 14, 15);

    // B fragments: 16 contiguous halves of codebook row n at K=k0+16*laneHi
    const _Float16* bBase = B + (size_t)(nBase + lane15) * D + k0 + 16 * laneHi;
    v16h b[4];
#pragma unroll
    for (int ni = 0; ni < 4; ++ni)
      b[ni] = *(const v16h*)(bBase + (size_t)(16 * ni) * D);

#pragma unroll
    for (int ni = 0; ni < 4; ++ni) {
      acc[0][ni] = __builtin_amdgcn_wmma_f32_16x16x32_f16(
          false, a0, false, b[ni], (short)0, acc[0][ni], false, false);
      acc[1][ni] = __builtin_amdgcn_wmma_f32_16x16x32_f16(
          false, a1, false, b[ni], (short)0, acc[1][ni], false, false);
    }
  }

  // C layout: lane15 = N, VGPR v -> M = v + 8*laneHi (+16 for second M tile)
#pragma unroll
  for (int mi = 0; mi < 2; ++mi) {
    float* cBase =
        C + (size_t)(mTile + 16 * mi + 8 * laneHi) * N_CB + nBase + lane15;
#pragma unroll
    for (int v = 0; v < 8; ++v) {
      float* r = cBase + (size_t)v * N_CB;
      r[0]  = acc[mi][0][v];
      r[16] = acc[mi][1][v];
      r[32] = acc[mi][2][v];
      r[48] = acc[mi][3][v];
    }
  }
}

// ---------------------------------------------------------------------------
// Per-row argmax over 8320 logits, gather z_q / z_q_st, loss accumulation.
// ---------------------------------------------------------------------------
__global__ __launch_bounds__(256)
void argmax_kernel(const float* __restrict__ logits,
                   const float* __restrict__ xn,
                   const float* __restrict__ cb,
                   float* __restrict__ outIdx,
                   float* __restrict__ zq,
                   float* __restrict__ zqst,
                   float* __restrict__ sums) {
  const int row = blockIdx.x;
  const int t   = threadIdx.x;
  const float* lr = logits + (size_t)row * N_CB;

  float bv = -3.4e38f;
  int   bi = 0;
  for (int i = t; i < N_CB; i += 256) {
    const float v = lr[i];
    if (v > bv) { bv = v; bi = i; }  // keeps first occurrence within stride
  }

  __shared__ float sv[256];
  __shared__ int   si[256];
  sv[t] = bv; si[t] = bi;
  __syncthreads();
  for (int off = 128; off > 0; off >>= 1) {
    if (t < off) {
      const float ov = sv[t + off];
      const int   oi = si[t + off];
      if (ov > sv[t] || (ov == sv[t] && oi < si[t])) { sv[t] = ov; si[t] = oi; }
    }
    __syncthreads();
  }
  const int idx = si[0];

  const float* cbr = cb + (size_t)idx * D;
  const float* xr  = xn + (size_t)row * D;
  float dp = 0.f, nx = 0.f, nz = 0.f;
#pragma unroll
  for (int i = 0; i < 4; ++i) {
    const int j = t + i * 256;
    const float zv = cbr[j];
    const float xv = xr[j];
    zq[(size_t)row * D + j]   = zv;
    zqst[(size_t)row * D + j] = zv;   // forward value of straight-through == z_q
    dp += xv * zv; nx += xv * xv; nz += zv * zv;
  }

  __shared__ float rd[256], rx[256], rz[256];
  rd[t] = dp; rx[t] = nx; rz[t] = nz;
  __syncthreads();
  for (int off = 128; off > 0; off >>= 1) {
    if (t < off) { rd[t] += rd[t + off]; rx[t] += rx[t + off]; rz[t] += rz[t + off]; }
    __syncthreads();
  }

  if (t == 0) {
    outIdx[row] = (float)idx;
    const float cosv = rd[0] / (fmaxf(sqrtf(rx[0]), EPSF) * fmaxf(sqrtf(rz[0]), EPSF));
    const float per  = 1.f - cosv;
    atomicAdd(&sums[0], per);                 // commitment sum
    if (idx >= N_SEM) {
      atomicAdd(&sums[1], per);               // vq sum
      atomicAdd(&sums[2], 1.0f);              // learnable count
    }
  }
}

__global__ void init_kernel(float* sums) {
  sums[0] = 0.f; sums[1] = 0.f; sums[2] = 0.f;
}

__global__ void fin_kernel(const float* __restrict__ sums, float* __restrict__ out) {
  const float commit = sums[0] / (float)N_X;
  const float vq     = sums[1] / (sums[2] + 1e-6f);
  out[0] = vq;                    // vq_loss
  out[1] = commit;                // commitment_loss
  out[2] = vq + 0.25f * commit;   // quant_loss
}

// ---------------------------------------------------------------------------
extern "C" void kernel_launch(void* const* d_in, const int* in_sizes, int n_in,
                              void* d_out, int out_size, void* d_ws, size_t ws_size,
                              hipStream_t stream) {
  const float* x   = (const float*)d_in[0];
  const float* sem = (const float*)d_in[1];
  const float* lrn = (const float*)d_in[2];
  float* out = (float*)d_out;

  // Workspace layout (all sizes 256B-aligned by construction)
  char* ws = (char*)d_ws;
  size_t off = 0;
  float* xn = (float*)(ws + off);        off += (size_t)N_X  * D * sizeof(float);
  float* cb = (float*)(ws + off);        off += (size_t)N_CB * D * sizeof(float);
  _Float16* xnh = (_Float16*)(ws + off); off += (size_t)N_X  * D * sizeof(_Float16);
  _Float16* cbh = (_Float16*)(ws + off); off += (size_t)N_CB * D * sizeof(_Float16);
  float* sums = (float*)(ws + off);

  // Output layout: logits | indices | z_q | z_q_st | vq | commit | quant
  float* logits = out;
  float* idxOut = out + (size_t)N_X * N_CB;
  float* zq     = idxOut + N_X;
  float* zqst   = zq + (size_t)N_X * D;
  float* scl    = zqst + (size_t)N_X * D;

  nrm_kernel<<<N_X, 256, 0, stream>>>(x, xn, xnh);
  nrm_kernel<<<N_SEM, 256, 0, stream>>>(sem, cb, cbh);
  nrm_kernel<<<N_LRN, 256, 0, stream>>>(lrn, cb + (size_t)N_SEM * D,
                                        cbh + (size_t)N_SEM * D);
  init_kernel<<<1, 1, 0, stream>>>(sums);

  // 8320/64 = 130 N-strips, 8192/(8 waves * 32 rows) = 32 M-blocks
  gemm_kernel<<<dim3(130, 32), 256, 0, stream>>>(xnh, cbh, logits);

  argmax_kernel<<<N_X, 256, 0, stream>>>(logits, xn, cb, idxOut, zq, zqst, sums);
  fin_kernel<<<1, 1, 0, stream>>>(sums, scl);
}
